// DynamicEdgeConv_13872744366780
// MI455X (gfx1250) — matrix operand order repaired and checked by reading, not compiled, not run on hardware
//
#include <hip/hip_runtime.h>

typedef float  v8f  __attribute__((ext_vector_type(8)));
typedef __bf16 v8bf __attribute__((ext_vector_type(8)));
typedef __bf16 v16bf __attribute__((ext_vector_type(16)));

#define NN 16384
#define FF 64
#define KK 16
#define HH 128
#define OO 128
#define CH 4                    // column chunks (parallelism for kNN)
#define NCT (NN / KK / CH)      // 256 column tiles per chunk
#define INFV 3.0e38f

static __device__ inline v16bf cat8(v8bf a, v8bf b) {
  return __builtin_shufflevector(a, b, 0, 1, 2, 3, 4, 5, 6, 7,
                                 8, 9, 10, 11, 12, 13, 14, 15);
}

#define WMMA_BF16(A, B, C) \
  __builtin_amdgcn_wmma_f32_16x16x32_bf16(false, (A), false, (B), (short)0, (C), false, false)

// ---------------------------------------------------------------- norms ----
__global__ __launch_bounds__(256) void norms_kernel(const float* __restrict__ x,
                                                    float* __restrict__ sq) {
  int i = blockIdx.x * blockDim.x + threadIdx.x;
  const float4* xv = (const float4*)(x + (size_t)i * FF);
  float s = 0.f;
#pragma unroll
  for (int t = 0; t < FF / 4; ++t) {
    float4 v = xv[t];
    s += v.x * v.x + v.y * v.y + v.z * v.z + v.w * v.w;
  }
  sq[i] = s;
}

// -------------------------------------------------- bf16 hi/lo splitting ----
// x = hi + lo with hi = bf16(x), lo = bf16(x - hi): ~16 mantissa bits total.
__global__ __launch_bounds__(256) void split_kernel(const float* __restrict__ x,
                                                    __bf16* __restrict__ xhi,
                                                    __bf16* __restrict__ xlo) {
  int t = blockIdx.x * blockDim.x + threadIdx.x;  // over N*F elements
  float v = x[t];
  __bf16 h = (__bf16)v;
  xhi[t] = h;
  xlo[t] = (__bf16)(v - (float)h);
}

// ----------------------------------------- fused split-bf16 Gram + top-K ----
// Block = 256 threads = 8 waves; wave owns one 16-row tile; block owns one
// column chunk. Gram tile = 6 chained V_WMMA_F32_16X16X32_BF16 (hh+hl+lh).
// Ballot fast-gate skips the serial insertion path for tiles with no winner.
__global__ __launch_bounds__(256) void knn_kernel(const __bf16* __restrict__ xhi,
                                                  const __bf16* __restrict__ xlo,
                                                  const float* __restrict__ sq,
                                                  float* __restrict__ candd,
                                                  int* __restrict__ candi) {
  __shared__ __align__(32) __bf16 xch[KK * FF];   // column tile hi (2KB)
  __shared__ __align__(32) __bf16 xcl[KK * FF];   // column tile lo (2KB)
  __shared__ float sqc[KK];
  __shared__ float dsc[8][KK][KK];                // slow-path spill (8KB)
  __shared__ __align__(64) float thr_s[8][KK];    // per-row 16th-best
  __shared__ float topd[128][KK];                 // running top-K dists
  __shared__ int   topi[128][KK];                 // running top-K ids

  const int tid  = threadIdx.x;
  const int wave = tid >> 5;
  const int lane = tid & 31;
  const int half = lane >> 4;
  const int l16  = lane & 15;
  const int chunk = blockIdx.x & (CH - 1);
  const int r0    = ((blockIdx.x >> 2) * 8 + wave) * 16;

  // A fragments (16x64 row tile, hi & lo), bf16 16x16x32 layout:
  // lane half h: elements 0..15 -> K = kb+16h .. +15 (contiguous)
  v16bf ah0, ah1, al0, al1;
  {
    const __bf16* rh = xhi + (size_t)(r0 + l16) * FF;
    const __bf16* rl = xlo + (size_t)(r0 + l16) * FF;
    const int kb = half * 16;
    ah0 = cat8(*(const v8bf*)(rh + kb),      *(const v8bf*)(rh + kb + 8));
    ah1 = cat8(*(const v8bf*)(rh + kb + 32), *(const v8bf*)(rh + kb + 40));
    al0 = cat8(*(const v8bf*)(rl + kb),      *(const v8bf*)(rl + kb + 8));
    al1 = cat8(*(const v8bf*)(rl + kb + 32), *(const v8bf*)(rl + kb + 40));
  }

  float curmax = INFV;
  int   maxpos = 0;
  const int lr = wave * 16 + l16;
  if (lane < 16) {
    thr_s[wave][l16] = INFV;
#pragma unroll
    for (int t = 0; t < KK; ++t) { topd[lr][t] = INFV; topi[lr][t] = 0; }
  }
  __syncthreads();

  for (int ct = 0; ct < NCT; ++ct) {
    const int c0 = (chunk * NCT + ct) * KK;
    // stage column tile (hi: threads 0-127, lo: threads 128-255)
    if (tid < 128)
      ((float4*)xch)[tid] = ((const float4*)(xhi + (size_t)c0 * FF))[tid];
    else
      ((float4*)xcl)[tid - 128] = ((const float4*)(xlo + (size_t)c0 * FF))[tid - 128];
    if (tid < KK) sqc[tid] = sq[c0 + tid];
    __syncthreads();

    if (ct + 1 < NCT) {                 // prefetch next tile (near scope)
      __builtin_prefetch(xhi + (size_t)(c0 + KK) * FF + lane * 64, 0, 3);
      __builtin_prefetch(xlo + (size_t)(c0 + KK) * FF + lane * 64, 0, 3);
    }

    // B fragments: lane half h holds col=l16, K = 32c + 16h .. +15 (contiguous)
    const __bf16* bch = xch + l16 * FF + half * 16;
    const __bf16* bcl = xcl + l16 * FF + half * 16;
    v16bf bh0 = *(const v16bf*)(bch);
    v16bf bh1 = *(const v16bf*)(bch + 32);
    v16bf bl0 = *(const v16bf*)(bcl);
    v16bf bl1 = *(const v16bf*)(bcl + 32);

    // dot = hi*hi + hi*lo + lo*hi  (f32 accumulate)
    v8f acc = {};
    acc = WMMA_BF16(ah0, bh0, acc);
    acc = WMMA_BF16(ah1, bh1, acc);
    acc = WMMA_BF16(ah0, bl0, acc);
    acc = WMMA_BF16(ah1, bl1, acc);
    acc = WMMA_BF16(al0, bh0, acc);
    acc = WMMA_BF16(al1, bh1, acc);

    // per-row thresholds for this lane's 8 rows (m = r + 8*half)
    const float4* tp = (const float4*)&thr_s[wave][half * 8];
    float4 t0 = tp[0], t1 = tp[1];
    const float th[8] = {t0.x, t0.y, t0.z, t0.w, t1.x, t1.y, t1.z, t1.w};

    const float sqn = sqc[l16];
    float v[8];
    unsigned hitmask = 0;
#pragma unroll
    for (int r = 0; r < 8; ++r) {
      const int m = r + half * 8;
      float d = fmaf(-2.0f, acc[r], sqn);       // sq_j - 2*dot (monotone)
      if (c0 + l16 == r0 + m) d = INFV;         // exclude self-loop
      v[r] = d;
      unsigned long long b = __ballot(d < th[r]);
      if (b & 0xffffu) hitmask |= 1u << r;
      if (b >> 16)     hitmask |= 1u << (r + 8);
    }

    if (hitmask) {                               // rare slow path
#pragma unroll
      for (int r = 0; r < 8; ++r) dsc[wave][r + half * 8][l16] = v[r];
      if (lane < 16 && ((hitmask >> l16) & 1u)) {
#pragma unroll 1
        for (int j = 0; j < KK; ++j) {
          float d = dsc[wave][l16][j];
          if (d < curmax) {
            topd[lr][maxpos] = d;
            topi[lr][maxpos] = c0 + j;
            float m0 = topd[lr][0];
            int   p0 = 0;
#pragma unroll
            for (int t = 1; t < KK; ++t) {
              float dt = topd[lr][t];
              if (dt > m0) { m0 = dt; p0 = t; }
            }
            curmax = m0; maxpos = p0;
          }
        }
        thr_s[wave][l16] = curmax;
      }
    }
    __syncthreads();
  }

  if (lane < 16) {
    const size_t base = ((size_t)chunk * NN + (r0 + l16)) * KK;
#pragma unroll
    for (int t = 0; t < KK; ++t) {
      candd[base + t] = topd[lr][t];
      candi[base + t] = topi[lr][t];
    }
  }
}

// ----------------------------------------------- merge per-chunk top-16 ----
__global__ __launch_bounds__(256) void merge_kernel(const float* __restrict__ candd,
                                                    const int* __restrict__ candi,
                                                    int* __restrict__ knn) {
  __shared__ float td[256][KK];
  __shared__ int   ti[256][KK];
  const int t   = threadIdx.x;
  const int row = blockIdx.x * 256 + t;
  float curmax = INFV;
  int   maxpos = 0;
#pragma unroll
  for (int q = 0; q < KK; ++q) { td[t][q] = INFV; ti[t][q] = 0; }
  for (int ch = 0; ch < CH; ++ch) {
    const size_t base = ((size_t)ch * NN + row) * KK;
#pragma unroll 1
    for (int q = 0; q < KK; ++q) {
      float d = candd[base + q];
      int  id = candi[base + q];
      if (d < curmax) {
        td[t][maxpos] = d; ti[t][maxpos] = id;
        float m0 = td[t][0]; int p0 = 0;
#pragma unroll
        for (int s = 1; s < KK; ++s) {
          float dt = td[t][s];
          if (dt > m0) { m0 = dt; p0 = s; }
        }
        curmax = m0; maxpos = p0;
      }
    }
  }
#pragma unroll
  for (int q = 0; q < KK; ++q) knn[(size_t)row * KK + q] = ti[t][q];
}

// --------------------------------------- layer-1 factorization GEMMs -------
// feat@W1 = x_i@W1a + (x_j - x_i)@W1b = R_i + Q_j  with
//   R = x @ (W1a - W1b) + b1   and   Q = x @ W1b.
// Two [N x 64] @ [64 x 128] split-bf16 GEMMs; weights split+transposed in LDS.
__global__ __launch_bounds__(256) void proj_kernel(const __bf16* __restrict__ xhi,
                                                   const __bf16* __restrict__ xlo,
                                                   const float* __restrict__ W1,
                                                   const float* __restrict__ b1,
                                                   float* __restrict__ R,
                                                   float* __restrict__ Q) {
  extern __shared__ float4 dynshm[];
  char* base = (char*)dynshm;
  __bf16* wRh = (__bf16*)base;            // [o][k] col-major, 64x128 -> 16KB
  __bf16* wRl = wRh + 8192;
  __bf16* wQh = wRl + 8192;
  __bf16* wQl = wQh + 8192;
  float*  b1s = (float*)(base + 65536);   // 512B

  const int tid  = threadIdx.x;
  const int wave = tid >> 5;
  const int lane = tid & 31;
  const int half = lane >> 4;
  const int l16  = lane & 15;
  const int r0   = (blockIdx.x * 8 + wave) * 16;

  for (int t = tid; t < 64 * HH; t += 256) {     // stage split weights
    const int k = t >> 7, o = t & 127;
    float a = W1[k * HH + o];
    float b = W1[(64 + k) * HH + o];
    float wr = a - b;
    __bf16 h1 = (__bf16)wr;
    wRh[o * FF + k] = h1;
    wRl[o * FF + k] = (__bf16)(wr - (float)h1);
    __bf16 h2 = (__bf16)b;
    wQh[o * FF + k] = h2;
    wQl[o * FF + k] = (__bf16)(b - (float)h2);
  }
  if (tid < HH) b1s[tid] = b1[tid];
  __syncthreads();

  v16bf ah0, ah1, al0, al1;                      // row-tile A fragments
  {
    const __bf16* rh = xhi + (size_t)(r0 + l16) * FF;
    const __bf16* rl = xlo + (size_t)(r0 + l16) * FF;
    const int kb = half * 16;
    ah0 = cat8(*(const v8bf*)(rh + kb),      *(const v8bf*)(rh + kb + 8));
    ah1 = cat8(*(const v8bf*)(rh + kb + 32), *(const v8bf*)(rh + kb + 40));
    al0 = cat8(*(const v8bf*)(rl + kb),      *(const v8bf*)(rl + kb + 8));
    al1 = cat8(*(const v8bf*)(rl + kb + 32), *(const v8bf*)(rl + kb + 40));
  }

  for (int ot = 0; ot < 8; ++ot) {
    const int o0 = ot * 16;
    const __bf16* pR_h = wRh + (o0 + l16) * FF + half * 16;
    const __bf16* pR_l = wRl + (o0 + l16) * FF + half * 16;
    const __bf16* pQ_h = wQh + (o0 + l16) * FF + half * 16;
    const __bf16* pQ_l = wQl + (o0 + l16) * FF + half * 16;

    v16bf bh0 = *(const v16bf*)(pR_h), bh1 = *(const v16bf*)(pR_h + 32);
    v16bf bl0 = *(const v16bf*)(pR_l), bl1 = *(const v16bf*)(pR_l + 32);
    v8f aR = {};
    aR = WMMA_BF16(ah0, bh0, aR); aR = WMMA_BF16(ah1, bh1, aR);
    aR = WMMA_BF16(ah0, bl0, aR); aR = WMMA_BF16(ah1, bl1, aR);
    aR = WMMA_BF16(al0, bh0, aR); aR = WMMA_BF16(al1, bh1, aR);

    v16bf ch0 = *(const v16bf*)(pQ_h), ch1 = *(const v16bf*)(pQ_h + 32);
    v16bf cl0 = *(const v16bf*)(pQ_l), cl1 = *(const v16bf*)(pQ_l + 32);
    v8f aQ = {};
    aQ = WMMA_BF16(ah0, ch0, aQ); aQ = WMMA_BF16(ah1, ch1, aQ);
    aQ = WMMA_BF16(ah0, cl0, aQ); aQ = WMMA_BF16(ah1, cl1, aQ);
    aQ = WMMA_BF16(al0, ch0, aQ); aQ = WMMA_BF16(al1, ch1, aQ);

    const float bb = b1s[o0 + l16];
#pragma unroll
    for (int r = 0; r < 8; ++r) {
      const size_t row = (size_t)(r0 + r + half * 8);
      R[row * HH + o0 + l16] = aR[r] + bb;
      Q[row * HH + o0 + l16] = aQ[r];
    }
  }
}

// ------------------------------ h = relu(R_i + Q_j); e = h@W2; max over j ---
// One node per wave; layer 2 = split-bf16 WMMA, W2 split+transposed in LDS.
__global__ __launch_bounds__(128) void mlp2_kernel(const float* __restrict__ R,
                                                   const float* __restrict__ Q,
                                                   const int* __restrict__ knn,
                                                   const float* __restrict__ W2,
                                                   const float* __restrict__ b2,
                                                   float* __restrict__ out) {
  extern __shared__ float4 dynshm[];
  char* base = (char*)dynshm;
  __bf16* w2h = (__bf16*)base;            // [o][k] col-major, 128x128 -> 32KB
  __bf16* w2l = w2h + 16384;              // 32KB
  float*  b2s = (float*)(base + 65536);   // 512B

  const int tid  = threadIdx.x;
  const int wave = tid >> 5;
  const int lane = tid & 31;
  const int half = lane >> 4;
  const int l16  = lane & 15;
  char* wbase = base + 66048 + wave * 8704;
  __bf16* hh = (__bf16*)wbase;            // 16x128 bf16 hi (4KB)
  __bf16* hl = hh + 2048;                 // 16x128 bf16 lo (4KB)
  int*    nbuf = (int*)(wbase + 8192);    // 16 neighbor ids

  for (int t = tid; t < HH * HH; t += 128) {   // stage split W2 (transposed)
    const int k = t >> 7, o = t & 127;
    float v = W2[k * HH + o];
    __bf16 h = (__bf16)v;
    w2h[o * HH + k] = h;
    w2l[o * HH + k] = (__bf16)(v - (float)h);
  }
  b2s[tid] = b2[tid];
  __syncthreads();

  const int node = blockIdx.x * 4 + wave;
  if (lane < 16) nbuf[lane] = knn[(size_t)node * KK + lane];

  // h[j][c] = relu(R[node][c] + Q[nb_j][c]); split to bf16 hi/lo in LDS
  float rn[4];
#pragma unroll
  for (int q = 0; q < 4; ++q) rn[q] = R[(size_t)node * HH + lane + 32 * q];
#pragma unroll 1
  for (int row = 0; row < KK; ++row) {
    const float* Qj = Q + (size_t)nbuf[row] * HH;
#pragma unroll
    for (int q = 0; q < 4; ++q) {
      float hv = rn[q] + Qj[lane + 32 * q];
      hv = hv > 0.f ? hv : 0.f;
      __bf16 h = (__bf16)hv;
      hh[row * HH + lane + 32 * q] = h;
      hl[row * HH + lane + 32 * q] = (__bf16)(hv - (float)h);
    }
  }

  const __bf16* ar_h = hh + l16 * HH + half * 16;
  const __bf16* ar_l = hl + l16 * HH + half * 16;
  for (int ot = 0; ot < 8; ++ot) {
    const int o0 = ot * 16;
    const __bf16* bc_h = w2h + (o0 + l16) * HH + half * 16;
    const __bf16* bc_l = w2l + (o0 + l16) * HH + half * 16;
    v8f acc = {};
#pragma unroll
    for (int c = 0; c < 4; ++c) {
      v16bf ah = cat8(*(const v8bf*)(ar_h + 32 * c), *(const v8bf*)(ar_h + 32 * c + 8));
      v16bf al = cat8(*(const v8bf*)(ar_l + 32 * c), *(const v8bf*)(ar_l + 32 * c + 8));
      v16bf bh = *(const v16bf*)(bc_h + 32 * c);
      v16bf bl = *(const v16bf*)(bc_l + 32 * c);
      acc = WMMA_BF16(ah, bh, acc);
      acc = WMMA_BF16(ah, bl, acc);
      acc = WMMA_BF16(al, bh, acc);
    }
    float m8 = acc[0];
#pragma unroll
    for (int r = 1; r < 8; ++r) m8 = fmaxf(m8, acc[r]);
    float other = __shfl_xor(m8, 16, 32);     // combine M=0..7 with M=8..15
    m8 = fmaxf(m8, other);
    if (lane < 16) out[(size_t)node * OO + o0 + lane] = m8 + b2s[o0 + lane];
  }
}

// ------------------------------------------------------------- launcher ----
extern "C" void kernel_launch(void* const* d_in, const int* in_sizes, int n_in,
                              void* d_out, int out_size, void* d_ws, size_t ws_size,
                              hipStream_t stream) {
  const float* x  = (const float*)d_in[0];
  const float* W1 = (const float*)d_in[1];
  const float* b1 = (const float*)d_in[2];
  const float* W2 = (const float*)d_in[3];
  const float* b2 = (const float*)d_in[4];
  float* out = (float*)d_out;

  char* ws = (char*)d_ws;
  float*  sq    = (float*)(ws);                                   // 64KB
  __bf16* xhi   = (__bf16*)(ws + (64 << 10));                     // 2MB
  __bf16* xlo   = (__bf16*)(ws + (64 << 10) + (2 << 20));         // 2MB
  float*  candd = (float*)(ws + (64 << 10) + (4 << 20));          // 4MB
  int*    candi = (int*)  (ws + (64 << 10) + (8 << 20));          // 4MB
  int*    knn   = (int*)  (ws + (64 << 10) + (12 << 20));         // 1MB
  float*  Rbuf  = (float*)(ws + (64 << 10) + (13 << 20));         // 8MB
  float*  Qbuf  = (float*)(ws + (64 << 10) + (21 << 20));         // 8MB

  norms_kernel<<<NN / 256, 256, 0, stream>>>(x, sq);
  split_kernel<<<NN * FF / 256, 256, 0, stream>>>(x, xhi, xlo);
  knn_kernel<<<(NN / 128) * CH, 256, 0, stream>>>(xhi, xlo, sq, candd, candi);
  merge_kernel<<<NN / 256, 256, 0, stream>>>(candd, candi, knn);
  proj_kernel<<<NN / 128, 256, 66048, stream>>>(xhi, xlo, W1, b1, Rbuf, Qbuf);
  mlp2_kernel<<<NN / 4, 128, 100864, stream>>>(Rbuf, Qbuf, knn, W2, b2, out);
}
